// MoELayer_25366076850383
// MI455X (gfx1250) — matrix operand order repaired and checked by reading, not compiled, not run on hardware
//
#include <hip/hip_runtime.h>
#include <cmath>

#define BQ 4
#define SQ 2048
#define DQ 1024
#define EQ 8
#define HQ 4096
#define TQ (BQ * SQ)   // 8192 tokens

#define MT 64          // tokens per workgroup (FFN kernel)
#define HC 32          // H-chunk (== WMMA K for GEMM2)

typedef __attribute__((ext_vector_type(16))) __bf16 v16bf;
typedef __attribute__((ext_vector_type(8)))  __bf16 v8bf;
typedef __attribute__((ext_vector_type(8)))  float  v8f;
typedef __attribute__((ext_vector_type(4)))  float  v4f;

static __device__ __forceinline__ __bf16 f2bf(float f) {
    // round-to-nearest-even f32 -> bf16
    unsigned u = __builtin_bit_cast(unsigned, f);
    u += 0x7fffu + ((u >> 16) & 1u);
    unsigned short h = (unsigned short)(u >> 16);
    return __builtin_bit_cast(__bf16, h);
}

static __device__ __forceinline__ v16bf cat8(v8bf lo, v8bf hi) {
    return __builtin_shufflevector(lo, hi, 0,1,2,3,4,5,6,7,8,9,10,11,12,13,14,15);
}

static __device__ __forceinline__ v8f vzero8() {
    v8f z;
#pragma unroll
    for (int i = 0; i < 8; ++i) z[i] = 0.0f;
    return z;
}

// LDS byte offset of a __shared__ object: generic LDS addresses carry the
// byte offset in the low 32 bits.
static __device__ __forceinline__ unsigned lds_off(const void* p) {
    return (unsigned)(uintptr_t)p;
}

// Async global->LDS copies (raw bytes, no conversion): CDNA5 ASYNCcnt path.
static __device__ __forceinline__ void async_ld_b128(unsigned lds, const void* g) {
    asm volatile("global_load_async_to_lds_b128 %0, %1, off"
                 :: "v"(lds), "v"((unsigned long long)(uintptr_t)g) : "memory");
}
static __device__ __forceinline__ void async_ld_b32(unsigned lds, const void* g) {
    asm volatile("global_load_async_to_lds_b32 %0, %1, off"
                 :: "v"(lds), "v"((unsigned long long)(uintptr_t)g) : "memory");
}
static __device__ __forceinline__ void wait_async0() {
    asm volatile("s_wait_asynccnt 0" ::: "memory");
}

// ---------------------------------------------------------------------------
// Router: logits = x @ router_w^T, softmax over E, top-2, per-(token,expert)
// coefficient written densely to coef[T][E] (zeros for unselected experts).
// One wave (32 lanes) per token.
// ---------------------------------------------------------------------------
__global__ __launch_bounds__(256) void moe_router(const float* __restrict__ x,
                                                  const float* __restrict__ rw,
                                                  float* __restrict__ coef) {
    __shared__ float rws[EQ][DQ] __attribute__((aligned(16)));   // 32 KB
    const int tid = threadIdx.x;

    // stage router weights via async global->LDS (no conversion needed)
    {
        const unsigned base = lds_off(&rws[0][0]);
#pragma unroll
        for (int i = 0; i < (EQ * DQ) / (256 * 4); ++i) {       // 8 x b128 per thread
            const int idx = i * 256 + tid;                       // float4 index
            async_ld_b128(base + (unsigned)idx * 16u, rw + (size_t)idx * 4);
        }
        wait_async0();
    }
    __syncthreads();

    const int wave = tid >> 5, lane = tid & 31;
    const int tok  = blockIdx.x * 8 + wave;

    float acc[EQ];
#pragma unroll
    for (int e = 0; e < EQ; ++e) acc[e] = 0.0f;

    const float* xrow = x + (size_t)tok * DQ;
    for (int i = lane; i < DQ; i += 32) {
        float xv = xrow[i];
#pragma unroll
        for (int e = 0; e < EQ; ++e) acc[e] += xv * rws[e][i];
    }
#pragma unroll
    for (int e = 0; e < EQ; ++e) {
#pragma unroll
        for (int off = 16; off > 0; off >>= 1)
            acc[e] += __shfl_xor(acc[e], off, 32);
    }
    // softmax over 8 logits (every lane redundantly)
    float m = acc[0];
#pragma unroll
    for (int e = 1; e < EQ; ++e) m = fmaxf(m, acc[e]);
    float p[EQ]; float s = 0.0f;
#pragma unroll
    for (int e = 0; e < EQ; ++e) { p[e] = __expf(acc[e] - m); s += p[e]; }
    const float inv = 1.0f / s;
    // top-2, first-index wins ties (matches lax.top_k)
    int i1 = 0; float v1 = p[0];
#pragma unroll
    for (int e = 1; e < EQ; ++e) if (p[e] > v1) { v1 = p[e]; i1 = e; }
    int i2 = -1; float v2 = -1.0f;
#pragma unroll
    for (int e = 0; e < EQ; ++e) if (e != i1 && p[e] > v2) { v2 = p[e]; i2 = e; }

    if (lane < EQ) {
        float c = (lane == i1) ? v1 * inv : ((lane == i2) ? v2 * inv : 0.0f);
        coef[(size_t)tok * EQ + lane] = c;
    }
}

// ---------------------------------------------------------------------------
// Fused MoE FFN: for each expert, out += coef .* (gelu(x @ w1^T) @ w2^T),
// computed densely (coef==0 kills unselected experts). Accumulation over
// experts AND H-chunks lives in one persistent f32 WMMA accumulator set,
// with the routing coefficient folded into the rows of h before GEMM2.
// Workgroup = 256 threads (8 wave32) owns 64 tokens x full D output.
// ---------------------------------------------------------------------------
__global__ __launch_bounds__(256) void moe_ffn(const float* __restrict__ x,
                                               const float* __restrict__ w1,
                                               const float* __restrict__ w2,
                                               const float* __restrict__ coef,
                                               float* __restrict__ out) {
    // LDS (bf16) ~273 KB of the 320 KB WGP budget; rows padded vs bank conflicts.
    __shared__ __bf16 xs [MT][DQ + 8] __attribute__((aligned(16)));   // x tile, [token][d]
    __shared__ __bf16 w1s[DQ][HC + 4] __attribute__((aligned(16)));   // w1 chunk, [d][h]
    __shared__ __bf16 w2s[HC][DQ + 8] __attribute__((aligned(16)));   // w2 chunk, [h][d]
    __shared__ __bf16 hs [MT][HC + 4] __attribute__((aligned(16)));   // h tile,  [token][h]
    __shared__ float  cs [MT] __attribute__((aligned(16)));           // routing coef, this expert

    const int tid  = threadIdx.x;
    const int wv   = tid >> 5;       // wave 0..7
    const int lane = tid & 31;
    const int l15  = lane & 15;
    const int hi   = lane >> 4;
    const int tok0 = blockIdx.x * MT;

    // ---- stage x tile once (f32 -> bf16) ----
    for (int r = 0; r < MT; ++r) {
        v4f v = *(const v4f*)(x + (size_t)(tok0 + r) * DQ + (size_t)tid * 4);
#pragma unroll
        for (int q = 0; q < 4; ++q) xs[r][tid * 4 + q] = f2bf(v[q]);
    }

    v8f acc[4][8];                    // 64 x 1024 f32 tile: wave wv owns cols [wv*128, wv*128+128)
#pragma unroll
    for (int mm = 0; mm < 4; ++mm)
#pragma unroll
        for (int j = 0; j < 8; ++j) acc[mm][j] = vzero8();

    for (int e = 0; e < EQ; ++e) {
        __syncthreads();              // also protects w2s/hs of previous expert's last chunk
        // routing coefficients for this expert: async global->LDS (f32, raw copy)
        if (tid < MT)
            async_ld_b32(lds_off(&cs[tid]), coef + (size_t)(tok0 + tid) * EQ + e);

        const float* w1e = w1 + (size_t)e * HQ * DQ;
        const float* w2e = w2 + (size_t)e * DQ * HQ;

        for (int hc = 0; hc < HQ / HC; ++hc) {
            if (hc) __syncthreads(); // previous GEMM2 done reading w2s/hs

            // ---- stage w1 chunk transposed: w1s[d][j] = w1e[(hc*HC+j)*DQ + d] ----
#pragma unroll 4
            for (int jj = 0; jj < HC; ++jj) {
                v4f v = *(const v4f*)(w1e + (size_t)(hc * HC + jj) * DQ + (size_t)tid * 4);
#pragma unroll
                for (int q = 0; q < 4; ++q) w1s[tid * 4 + q][jj] = f2bf(v[q]);
            }
            // ---- stage w2 chunk transposed: w2s[j][d] = w2e[d*HQ + hc*HC + j] ----
            {
                const int jq = (tid & 7) * 4;   // j base (0..28)
                const int dl = tid >> 3;        // 0..31
#pragma unroll 4
                for (int pass = 0; pass < DQ / 32; ++pass) {
                    int d = pass * 32 + dl;
                    v4f v = *(const v4f*)(w2e + (size_t)d * HQ + (size_t)(hc * HC) + jq);
#pragma unroll
                    for (int q = 0; q < 4; ++q) w2s[jq + q][d] = f2bf(v[q]);
                }
            }
            // ---- prefetch next chunk into L2 (global_prefetch_b8) ----
            if (hc + 1 < HQ / HC) {
                __builtin_prefetch(w1e + (size_t)((hc + 1) * HC + (tid >> 3)) * DQ +
                                       (size_t)(tid & 7) * 128, 0, 1);
#pragma unroll
                for (int q = 0; q < 4; ++q)
                    __builtin_prefetch(w2e + (size_t)(tid * 4 + q) * HQ +
                                           (size_t)((hc + 1) * HC), 0, 1);
            }
            wait_async0();           // cs landed (no-op except on expert entry)
            __syncthreads();

            // ---- GEMM1: h[64 x 32] = x_tile @ w1_chunk^T ; one 16x16 tile per wave ----
            {
                const int m1 = wv & 3, n1 = wv >> 2;
                v8f hacc = vzero8();
                const int arow = m1 * 16 + l15;
#pragma unroll 4
                for (int kb = 0; kb < DQ; kb += 32) {
                    const int ka = kb + 8 * hi;
                    v8bf alo = *(const v8bf*)&xs[arow][ka];
                    v8bf ahi = *(const v8bf*)&xs[arow][ka + 16];
                    v16bf a  = cat8(alo, ahi);
                    v16bf b;
#pragma unroll
                    for (int s = 0; s < 16; ++s)
                        b[s] = w1s[kb + ((s < 8) ? (8 * hi + s) : (16 + 8 * hi + s - 8))][n1 * 16 + l15];
                    hacc = __builtin_amdgcn_wmma_f32_16x16x32_bf16(
                        false, a, false, b, (short)0, hacc, false, false);
                }
                // exact GELU + routing-coef scale, store bf16 h tile
#pragma unroll
                for (int r = 0; r < 8; ++r) {
                    const int trow = m1 * 16 + r + 8 * hi;
                    float v = hacc[r];
                    float g = 0.5f * v * (1.0f + erff(v * 0.70710678f));
                    g *= cs[trow];
                    hs[trow][n1 * 16 + l15] = f2bf(g);
                }
            }
            __syncthreads();

            // ---- GEMM2: acc += h_tile @ w2_chunk^T (K = 32) ----
            {
                v16bf ha[4];
#pragma unroll
                for (int mm = 0; mm < 4; ++mm) {
                    const int arow = mm * 16 + l15;
                    const int ka = 8 * hi;
                    v8bf alo = *(const v8bf*)&hs[arow][ka];
                    v8bf ahi = *(const v8bf*)&hs[arow][ka + 16];
                    ha[mm] = cat8(alo, ahi);
                }
#pragma unroll
                for (int j = 0; j < 8; ++j) {
                    const int d0 = wv * 128 + j * 16;
                    v16bf b;
#pragma unroll
                    for (int s = 0; s < 16; ++s)
                        b[s] = w2s[(s < 8) ? (8 * hi + s) : (16 + 8 * hi + s - 8)][d0 + l15];
#pragma unroll
                    for (int mm = 0; mm < 4; ++mm)
                        acc[mm][j] = __builtin_amdgcn_wmma_f32_16x16x32_bf16(
                            false, ha[mm], false, b, (short)0, acc[mm][j], false, false);
                }
            }
        } // hc
    } // e

    // ---- write out tile (f32), each element exactly once ----
#pragma unroll
    for (int mm = 0; mm < 4; ++mm)
#pragma unroll
        for (int j = 0; j < 8; ++j) {
            const int d = wv * 128 + j * 16 + l15;
#pragma unroll
            for (int r = 0; r < 8; ++r) {
                const int trow = tok0 + mm * 16 + r + 8 * hi;
                out[(size_t)trow * DQ + d] = acc[mm][j][r];
            }
        }
}

// ---------------------------------------------------------------------------
extern "C" void kernel_launch(void* const* d_in, const int* in_sizes, int n_in,
                              void* d_out, int out_size, void* d_ws, size_t ws_size,
                              hipStream_t stream) {
    const float* x  = (const float*)d_in[0];   // [B,S,D]
    const float* rw = (const float*)d_in[1];   // [E,D]
    const float* w1 = (const float*)d_in[2];   // [E,H,D]
    const float* w2 = (const float*)d_in[3];   // [E,D,H]
    float* out  = (float*)d_out;               // [B,S,D]
    float* coef = (float*)d_ws;                // [T,E] = 256 KB scratch

    moe_router<<<TQ / 8, 256, 0, stream>>>(x, rw, coef);
    moe_ffn<<<TQ / MT, 256, 0, stream>>>(x, w1, w2, coef, out);
}
